// MegaNeRF_53206054863590
// MI455X (gfx1250) — matrix operand order repaired and checked by reading, not compiled, not run on hardware
//
#include <hip/hip_runtime.h>

typedef __attribute__((ext_vector_type(16))) _Float16 v16h;
typedef __attribute__((ext_vector_type(8)))  _Float16 v8h;
typedef __attribute__((ext_vector_type(8)))  float    v8f;

#define NPTS   65536
#define NEXP   8
#define DIN    90
#define HID    256
#define DOUT   4
#define XSTR   93          // 3 + DIN
#define LDH    264         // LDS row stride in halves (mult of 8 -> 16B aligned rows)

// ---- workspace layout (bytes) ----
static constexpr size_t OFF_ROUTE = 0;                               // N*E f32      = 2,097,152
static constexpr size_t OFF_FA    = OFF_ROUTE + (size_t)NPTS*NEXP*4; // 4096*3*512 h = 12,582,912
static constexpr size_t OFF_W1    = OFF_FA + (size_t)4096*3*512*2;   // 8*16*3*512 h =   393,216
static constexpr size_t OFF_W2    = OFF_W1 + (size_t)8*16*3*512*2;   // 8*16*8*512 h = 1,048,576
static constexpr size_t OFF_W3    = OFF_W2 + (size_t)8*16*8*512*2;   // 8*1*8*512 h  =    65,536
// total ~16.2 MB

__device__ __forceinline__ v8f wmma16(v16h a, v16h b, v8f c) {
    return __builtin_amdgcn_wmma_f32_16x16x32_f16(false, a, false, b, (short)0, c, false, false);
}

// A-fragment k mapping (16-bit A 16x32, ISA 7.12.2): lane hi half offsets
__device__ __forceinline__ int k_localA(int hi, int j) {
    return hi * 8 + j + ((j >= 8) ? 8 : 0);   // hi=0: 0..7,16..23 ; hi=1: 8..15,24..31
}

// ---------------- prep kernels ----------------

__global__ void routing_kernel(const float* __restrict__ x,
                               const float* __restrict__ cen,
                               float* __restrict__ w) {
    int n = blockIdx.x * blockDim.x + threadIdx.x;
    if (n >= NPTS) return;
    float px = x[(size_t)n * XSTR + 0];
    float py = x[(size_t)n * XSTR + 1];
    float pz = x[(size_t)n * XSTR + 2];
    float d[NEXP], dmin = 3.0e38f;
    #pragma unroll
    for (int e = 0; e < NEXP; e++) {
        float dx = px - cen[e * 3 + 0];
        float dy = py - cen[e * 3 + 1];
        float dz = pz - cen[e * 3 + 2];
        d[e] = sqrtf(dx * dx + dy * dy + dz * dz);
        dmin = fminf(dmin, d[e]);
    }
    float inv[NEXP], s = 0.f;
    #pragma unroll
    for (int e = 0; e < NEXP; e++) {
        float iv = 1.0f / (d[e] + 1e-8f);
        if (d[e] > 2.0f * dmin) iv = 0.0f;
        inv[e] = iv; s += iv;
    }
    float rs = 1.0f / s;
    #pragma unroll
    for (int e = 0; e < NEXP; e++) w[(size_t)n * NEXP + e] = inv[e] * rs;
}

// feats -> f16, pre-swizzled into WMMA A-fragment layout, K padded 90->96
// one thread per (tile16, kk, lane); writes 16 contiguous halves (32B)
__global__ void fragA_kernel(const float* __restrict__ x, _Float16* __restrict__ fa) {
    int gid = blockIdx.x * blockDim.x + threadIdx.x;     // 4096*3*32 = 393216
    if (gid >= 4096 * 3 * 32) return;
    int lane = gid & 31;
    int kk   = (gid >> 5) % 3;
    int tile = gid / 96;
    int hi = lane >> 4, lo = lane & 15;
    int row = tile * 16 + lo;
    v16h o;
    #pragma unroll
    for (int j = 0; j < 16; j++) {
        int k = kk * 32 + k_localA(hi, j);
        float v = (k < DIN) ? x[(size_t)row * XSTR + 3 + k] : 0.0f;
        o[j] = (_Float16)v;
    }
    *(v16h*)(fa + (size_t)gid * 16) = o;
}

// weights [E][Kreal][Nreal] -> f16 B-fragments (32x16 B tile: lanes0-15 K=0..7,16..23,
// lanes16-31 K=8..15,24..31; 16 contiguous halves per lane). Zero-pads K and N.
__global__ void fragB_kernel(const float* __restrict__ W, _Float16* __restrict__ fb,
                             int Kreal, int kchunks, int Nreal, int ntiles) {
    int gid = blockIdx.x * blockDim.x + threadIdx.x;     // E*ntiles*kchunks*32
    int total = NEXP * ntiles * kchunks * 32;
    if (gid >= total) return;
    int lane = gid & 31;
    int t = gid >> 5;
    int kk = t % kchunks; t /= kchunks;
    int ct = t % ntiles;  int e = t / ntiles;
    int n = ct * 16 + (lane & 15);
    v16h o;
    #pragma unroll
    for (int j = 0; j < 16; j++) {
        int k = kk * 32 + (lane >> 4) * 8 + j + ((j >= 8) ? 8 : 0);
        float v = (k < Kreal && n < Nreal) ? W[((size_t)e * Kreal + k) * (size_t)Nreal + n] : 0.0f;
        o[j] = (_Float16)v;
    }
    *(v16h*)(fb + (size_t)gid * 16) = o;
}

// ---------------- main MoE kernel ----------------

__device__ __forceinline__ v16h load_a_lds(const _Float16* h, int rowbase, int kk,
                                           int lo, int hi) {
    const _Float16* p = h + (size_t)(rowbase + lo) * LDH + kk * 32 + hi * 8;
    v8h x0 = *(const v8h*)p;          // K = kk*32 + hi*8 .. +7
    v8h x1 = *(const v8h*)(p + 16);   // K = kk*32 + 16 + hi*8 .. +7
    v16h a;
    #pragma unroll
    for (int j = 0; j < 8; j++) { a[j] = x0[j]; a[j + 8] = x1[j]; }
    return a;
}

__global__ __launch_bounds__(256, 1)
void moe_main(const _Float16* __restrict__ fa,
              const _Float16* __restrict__ fw1,
              const _Float16* __restrict__ fw2,
              const _Float16* __restrict__ fw3,
              const float* __restrict__ b1,
              const float* __restrict__ b2,
              const float* __restrict__ b3,
              const float* __restrict__ wroute,
              float* __restrict__ out) {
    __shared__ __align__(16) _Float16 h1[32 * LDH];
    __shared__ __align__(16) _Float16 h2[32 * LDH];

    const int lane = threadIdx.x & 31;
    const int wave = threadIdx.x >> 5;
    const int hi = lane >> 4;
    const int lo = lane & 15;
    const int blk = blockIdx.x;          // 2048 blocks * 32 rows

    // layer-1 A fragments (feats), shared by all experts: 2 row tiles x 3 k-chunks
    v16h a1[2][3];
    #pragma unroll
    for (int r = 0; r < 2; r++)
        #pragma unroll
        for (int kk = 0; kk < 3; kk++)
            a1[r][kk] = *(const v16h*)(fa + ((size_t)((blk * 2 + r) * 3 + kk) * 32 + lane) * 16);

    float outacc[8];
    #pragma unroll
    for (int j = 0; j < 8; j++) outacc[j] = 0.0f;

    const int ct0 = wave * 2;            // this wave's two column tiles
    const int ct1 = wave * 2 + 1;

    for (int e = 0; e < NEXP; e++) {
        // ---------- Layer 1: feats(90) -> h1(256), relu ----------
        {
            v8f c00 = {}, c01 = {}, c10 = {}, c11 = {};   // c[t][r]
            #pragma unroll
            for (int kk = 0; kk < 3; kk++) {
                v16h bb0 = *(const v16h*)(fw1 + ((size_t)((e * 16 + ct0) * 3 + kk) * 32 + lane) * 16);
                v16h bb1 = *(const v16h*)(fw1 + ((size_t)((e * 16 + ct1) * 3 + kk) * 32 + lane) * 16);
                c00 = wmma16(a1[0][kk], bb0, c00);
                c01 = wmma16(a1[1][kk], bb0, c01);
                c10 = wmma16(a1[0][kk], bb1, c10);
                c11 = wmma16(a1[1][kk], bb1, c11);
            }
            float bias0 = b1[e * HID + ct0 * 16 + lo];
            float bias1 = b1[e * HID + ct1 * 16 + lo];
            #pragma unroll
            for (int j = 0; j < 8; j++) {
                int m = hi * 8 + j;
                h1[(size_t)m        * LDH + ct0 * 16 + lo] = (_Float16)fmaxf(c00[j] + bias0, 0.0f);
                h1[(size_t)(16 + m) * LDH + ct0 * 16 + lo] = (_Float16)fmaxf(c01[j] + bias0, 0.0f);
                h1[(size_t)m        * LDH + ct1 * 16 + lo] = (_Float16)fmaxf(c10[j] + bias1, 0.0f);
                h1[(size_t)(16 + m) * LDH + ct1 * 16 + lo] = (_Float16)fmaxf(c11[j] + bias1, 0.0f);
            }
        }
        __syncthreads();
        // ---------- Layer 2: h1(256) -> h2(256), relu ----------
        {
            v8f c00 = {}, c01 = {}, c10 = {}, c11 = {};
            #pragma unroll
            for (int kk = 0; kk < 8; kk++) {
                v16h aA = load_a_lds(h1, 0, kk, lo, hi);
                v16h aB = load_a_lds(h1, 16, kk, lo, hi);
                v16h bb0 = *(const v16h*)(fw2 + ((size_t)((e * 16 + ct0) * 8 + kk) * 32 + lane) * 16);
                v16h bb1 = *(const v16h*)(fw2 + ((size_t)((e * 16 + ct1) * 8 + kk) * 32 + lane) * 16);
                c00 = wmma16(aA, bb0, c00);
                c01 = wmma16(aB, bb0, c01);
                c10 = wmma16(aA, bb1, c10);
                c11 = wmma16(aB, bb1, c11);
            }
            float bias0 = b2[e * HID + ct0 * 16 + lo];
            float bias1 = b2[e * HID + ct1 * 16 + lo];
            #pragma unroll
            for (int j = 0; j < 8; j++) {
                int m = hi * 8 + j;
                h2[(size_t)m        * LDH + ct0 * 16 + lo] = (_Float16)fmaxf(c00[j] + bias0, 0.0f);
                h2[(size_t)(16 + m) * LDH + ct0 * 16 + lo] = (_Float16)fmaxf(c01[j] + bias0, 0.0f);
                h2[(size_t)m        * LDH + ct1 * 16 + lo] = (_Float16)fmaxf(c10[j] + bias1, 0.0f);
                h2[(size_t)(16 + m) * LDH + ct1 * 16 + lo] = (_Float16)fmaxf(c11[j] + bias1, 0.0f);
            }
        }
        __syncthreads();
        // ---------- Layer 3: h2(256) -> out(4, padded 16), weighted blend ----------
        if (wave < 2) {                      // wave-uniform branch: EXEC stays full
            int r = wave;                    // row tile
            v8f c = {};
            #pragma unroll
            for (int kk = 0; kk < 8; kk++) {
                v16h a = load_a_lds(h2, r * 16, kk, lo, hi);
                v16h b = *(const v16h*)(fw3 + ((size_t)(e * 8 + kk) * 32 + lane) * 16);
                c = wmma16(a, b, c);
            }
            float bias = (lo < DOUT) ? b3[e * DOUT + lo] : 0.0f;
            #pragma unroll
            for (int j = 0; j < 8; j++) {
                int m = hi * 8 + j;
                int row = blk * 32 + r * 16 + m;
                float wgt = wroute[(size_t)row * NEXP + e];
                outacc[j] += wgt * (c[j] + bias);
            }
        }
        __syncthreads();
    }

    if (wave < 2 && lo < DOUT) {
        #pragma unroll
        for (int j = 0; j < 8; j++) {
            int m = hi * 8 + j;
            int row = blk * 32 + wave * 16 + m;
            out[(size_t)row * DOUT + lo] = outacc[j];
        }
    }
}

// ---------------- launcher ----------------

extern "C" void kernel_launch(void* const* d_in, const int* in_sizes, int n_in,
                              void* d_out, int out_size, void* d_ws, size_t ws_size,
                              hipStream_t stream) {
    const float* x   = (const float*)d_in[0];
    const float* cen = (const float*)d_in[1];
    const float* W1  = (const float*)d_in[2];
    const float* b1  = (const float*)d_in[3];
    const float* W2  = (const float*)d_in[4];
    const float* b2  = (const float*)d_in[5];
    const float* W3  = (const float*)d_in[6];
    const float* b3  = (const float*)d_in[7];
    float* out = (float*)d_out;

    char* ws = (char*)d_ws;
    float*    wroute = (float*)(ws + OFF_ROUTE);
    _Float16* fa     = (_Float16*)(ws + OFF_FA);
    _Float16* fw1    = (_Float16*)(ws + OFF_W1);
    _Float16* fw2    = (_Float16*)(ws + OFF_W2);
    _Float16* fw3    = (_Float16*)(ws + OFF_W3);

    routing_kernel<<<NPTS / 256, 256, 0, stream>>>(x, cen, wroute);
    fragA_kernel<<<(4096 * 3 * 32) / 256, 256, 0, stream>>>(x, fa);
    fragB_kernel<<<(NEXP * 16 * 3 * 32) / 256, 256, 0, stream>>>(W1, fw1, DIN, 3, HID, 16);
    fragB_kernel<<<(NEXP * 16 * 8 * 32) / 256, 256, 0, stream>>>(W2, fw2, HID, 8, HID, 16);
    fragB_kernel<<<(NEXP * 1 * 8 * 32) / 256, 256, 0, stream>>>(W3, fw3, HID, 8, DOUT, 1);
    moe_main<<<NPTS / 32, 256, 0, stream>>>(fa, fw1, fw2, fw3, b1, b2, b3, wroute, out);
}
